// CategorySpecificLinear_4741643895621
// MI455X (gfx1250) — matrix operand order repaired and checked by reading, not compiled, not run on hardware
//
#include <hip/hip_runtime.h>

typedef __attribute__((ext_vector_type(16))) _Float16 v16h;
typedef __attribute__((ext_vector_type(8)))  _Float16 v8h;
typedef __attribute__((ext_vector_type(4)))  _Float16 v4h;
typedef __attribute__((ext_vector_type(8)))  float    v8f;
typedef __attribute__((ext_vector_type(4)))  float    v4f;

#define NUM_CATS 16
#define DIN 1024
#define DH  4096
#define SEQ 512
#define NB  64

#define BM 128
#define BN 128
#define BK 32
#define BKP 40   // padded K stride in halves (80 B) to spread LDS banks

// out[b] = x[b] @ W[cat[b]] + bias[cat[b]]
// x: [64,512,1024] f32, W: [16,1024,4096] f32, bias: [16,4096] f32

struct StageCtx {
    const float* xb;   // x tile base (row-major, stride DIN)
    const float* Wc;   // W tile base (row-major, stride DH)
    int aRow, aKc;     // this thread's A staging assignment
    int bN, bKg;       // this thread's B staging assignment
};

// Convert one 128x32 A tile (f32->f16, [row][k]) and one 32x128 B tile
// (f32->f16, transposed into [n][k]) into the given LDS buffers.
__device__ __forceinline__ void stage_tiles(const StageCtx& s, int k0,
                                            _Float16* __restrict__ Asb,
                                            _Float16* __restrict__ Bsb)
{
    #pragma unroll
    for (int i = 0; i < 4; ++i) {
        const int row = s.aRow + i * 32;
        v4f v = *(const v4f*)(s.xb + (size_t)row * DIN + k0 + s.aKc);
        v4h h;
        h.x = (_Float16)v.x; h.y = (_Float16)v.y;
        h.z = (_Float16)v.z; h.w = (_Float16)v.w;
        *(v4h*)(&Asb[row * BKP + s.aKc]) = h;
    }
    #pragma unroll
    for (int i = 0; i < 4; ++i) {
        const int kk = s.bKg + i * 4;
        const float* wp = s.Wc + (size_t)(k0 + kk) * DH + s.bN;
        v4h h;
        h.x = (_Float16)wp[0];
        h.y = (_Float16)wp[DH];
        h.z = (_Float16)wp[2 * (size_t)DH];
        h.w = (_Float16)wp[3 * (size_t)DH];
        *(v4h*)(&Bsb[s.bN * BKP + kk]) = h;
    }
}

__global__ __launch_bounds__(256) void cat_linear_wmma(
    const float* __restrict__ x,
    const int*   __restrict__ cat_ids,
    const float* __restrict__ W,
    const float* __restrict__ bias,
    float* __restrict__ out)
{
    // double-buffered tiles: 2 * (10KiB + 10KiB) = 40 KiB of the 320 KiB WGP LDS
    __shared__ __attribute__((aligned(32))) _Float16 As[2][BM * BKP]; // [row][k]
    __shared__ __attribute__((aligned(32))) _Float16 Bs[2][BN * BKP]; // [n][k]

    const int nTile = blockIdx.x;          // 0..31
    const int mTile = blockIdx.y;          // 0..3
    const int bb    = blockIdx.z;          // 0..63
    const int c     = cat_ids[bb] & (NUM_CATS - 1);

    const float* xb = x    + (size_t)bb * SEQ * DIN + (size_t)mTile * BM * DIN;
    const float* Wc = W    + (size_t)c  * DIN * DH  + (size_t)nTile * BN;
    const float* bc = bias + (size_t)c  * DH        + (size_t)nTile * BN;
    float*       ob = out  + (size_t)bb * SEQ * DH  + (size_t)mTile * BM * DH
                           + (size_t)nTile * BN;

    const int t     = threadIdx.x;   // 0..255 (8 waves of 32)
    const int lane  = t & 31;
    const int wave  = t >> 5;
    const int waveM = wave >> 1;     // 0..3 -> rows  waveM*32
    const int waveN = wave & 1;      // 0..1 -> cols  waveN*64
    const int l16   = lane & 15;
    const int lHi   = lane >> 4;     // 0 or 1

    v8f acc[2][4];
    #pragma unroll
    for (int mf = 0; mf < 2; ++mf)
        #pragma unroll
        for (int nf = 0; nf < 4; ++nf)
            #pragma unroll
            for (int e = 0; e < 8; ++e)
                acc[mf][nf][e] = 0.0f;

    StageCtx s;
    s.xb = xb; s.Wc = Wc;
    s.aRow = t >> 3;          // 0..31 (4 rows each, stride 32)
    s.aKc  = (t & 7) << 2;    // 0,4,...,28
    s.bN   = t & 127;         // 0..127
    s.bKg  = (t >> 7) << 4;   // 0 or 16

    // prologue: fill buffer 0 with the first K-tile
    stage_tiles(s, 0, As[0], Bs[0]);
    __syncthreads();

    int buf = 0;
    for (int k0 = 0; k0 < DIN; k0 += BK) {
        // ---- stage NEXT K-tile into the other buffer (overlaps with WMMA below) ----
        if (k0 + BK < DIN)
            stage_tiles(s, k0 + BK, As[buf ^ 1], Bs[buf ^ 1]);
        if (k0 + 2 * BK < DIN) { // pull K-tile after that toward L2 (global_prefetch_b8)
            __builtin_prefetch(s.xb + (size_t)s.aRow * DIN + (k0 + 2 * BK) + s.aKc, 0, 1);
            __builtin_prefetch(s.Wc + (size_t)(k0 + 2 * BK + s.bKg) * DH + s.bN, 0, 1);
        }

        // ---- build fragments from CURRENT buffer per ISA VGPR layouts ----
        const _Float16* Acur = As[buf];
        const _Float16* Bcur = Bs[buf];

        // A 16x32 f16: lane M=lane%16; halves K = {kg*8..+7, kg*8+16..+23}, kg=lane/16
        v16h afrag[2];
        #pragma unroll
        for (int mf = 0; mf < 2; ++mf) {
            const int r  = waveM * 32 + mf * 16 + l16;
            const int kb = lHi * 8;
            v8h lo = *(const v8h*)(&Acur[r * BKP + kb]);
            v8h hi = *(const v8h*)(&Acur[r * BKP + kb + 16]);
            v16h a;
            #pragma unroll
            for (int e = 0; e < 8; ++e) { a[e] = lo[e]; a[e + 8] = hi[e]; }
            afrag[mf] = a;
        }
        // B 32x16 f16: lane N=lane%16; halves K = {kg*8..+7, kg*8+16..+23}, kg=lane/16
        v16h bfrag[4];
        #pragma unroll
        for (int nf = 0; nf < 4; ++nf) {
            const int n  = waveN * 64 + nf * 16 + l16;
            const int kb = lHi * 8;
            v8h lo = *(const v8h*)(&Bcur[n * BKP + kb]);
            v8h hi = *(const v8h*)(&Bcur[n * BKP + kb + 16]);
            v16h b;
            #pragma unroll
            for (int e = 0; e < 8; ++e) { b[e] = lo[e]; b[e + 8] = hi[e]; }
            bfrag[nf] = b;
        }

        // ---- 8 WMMAs per wave per K-step ----
        #pragma unroll
        for (int mf = 0; mf < 2; ++mf)
            #pragma unroll
            for (int nf = 0; nf < 4; ++nf)
                acc[mf][nf] = __builtin_amdgcn_wmma_f32_16x16x32_f16(
                    false, afrag[mf], false, bfrag[nf],
                    (short)0, acc[mf][nf], false, false);

        // one barrier per iteration: next-tile stores visible AND this-tile
        // reads complete before the buffer roles swap
        __syncthreads();
        buf ^= 1;
    }

    // ---- epilogue: C/D layout lane = col (N=lane%16), VGPR e = row (+8 for lanes 16..31) ----
    #pragma unroll
    for (int nf = 0; nf < 4; ++nf) {
        const int col = waveN * 64 + nf * 16 + l16;
        const float bv = bc[col];
        #pragma unroll
        for (int mf = 0; mf < 2; ++mf) {
            const int row0 = waveM * 32 + mf * 16 + lHi * 8;
            #pragma unroll
            for (int e = 0; e < 8; ++e)
                ob[(size_t)(row0 + e) * DH + col] = acc[mf][nf][e] + bv;
        }
    }
}

extern "C" void kernel_launch(void* const* d_in, const int* in_sizes, int n_in,
                              void* d_out, int out_size, void* d_ws, size_t ws_size,
                              hipStream_t stream) {
    (void)in_sizes; (void)n_in; (void)out_size; (void)d_ws; (void)ws_size;
    const float* x    = (const float*)d_in[0];
    const int*   cat  = (const int*)  d_in[1];
    const float* W    = (const float*)d_in[2];
    const float* bias = (const float*)d_in[3];
    float*       out  = (float*)d_out;

    dim3 grid(DH / BN, SEQ / BM, NB);   // (32, 4, 64)
    cat_linear_wmma<<<grid, 256, 0, stream>>>(x, cat, W, bias, out);
}
